// ConvAttn2_38723425141453
// MI455X (gfx1250) — matrix operand order repaired and checked by reading, not compiled, not run on hardware
//
#include <hip/hip_runtime.h>

typedef __attribute__((ext_vector_type(16))) _Float16 v16h;
typedef __attribute__((ext_vector_type(8)))  _Float16 v8h;
typedef __attribute__((ext_vector_type(8)))  float    v8f;

#define B_     16
#define NFEAT  16
#define TLEN   64
#define SCALE_ 32
#define HID_   128
#define NH_    2
#define DH_    64
#define H1Y    61
#define H1X    29
#define H2Y    58
#define H2X    26
#define L_     1508          // 58*26
#define U_     40
#define ML_    (B_*L_)       // 24128 = 377*64
#define K2_    2048          // 128*4*4 (conv2 reduction dim)

// ---------------- WMMA fragment loaders (contiguous k = 16h + s order in A and B) -----

// A: 16x32 f16 from a row-major LDS tile (stride in halves; stride*2 and the
// row base must be 16B aligned so each 8-half run is one ds_load_b128).
// lane 0-15:  halves 0-15 -> K=0..15   (M = lane)
// lane 16-31: halves 0-15 -> K=16..31  (M = lane-16)
__device__ inline v16h load_a_frag(const _Float16* p, int stride, int lane) {
  int m  = lane & 15;
  int kb = (lane & 16) ? 16 : 0;
  const _Float16* rp = p + m * stride + kb;
  v8h lo = *(const v8h*)rp;
  v8h hi = *(const v8h*)(rp + 8);
  return __builtin_shufflevector(lo, hi, 0, 1, 2, 3, 4, 5, 6, 7,
                                 8, 9, 10, 11, 12, 13, 14, 15);
}

// B: fragment-order LDS tile: frag[((nb*32)+lane)*16 + j] holds
// element (K = (lane>=16?16:0) + j, N = nb*16 + (lane&15)).  One v16h load.
__device__ inline v16h load_b_frag(const _Float16* fragBase, int nb, int lane) {
  return *(const v16h*)(fragBase + (nb * 32 + lane) * 16);
}

// C/D f32 16x16: VGPR r -> M = r + (lane>=16 ? 8 : 0), N = lane&15.

// ---------------- K0: weight prep (f32 -> f16, conv2_w transposed) ---------
__global__ void prep_weights(const float* __restrict__ w2, const float* __restrict__ wq,
                             const float* __restrict__ wk, const float* __restrict__ wv,
                             const float* __restrict__ wo,
                             _Float16* w2t, _Float16* wqh, _Float16* wkh,
                             _Float16* wvh, _Float16* woh) {
  int id = blockIdx.x * blockDim.x + threadIdx.x;   // 0 .. 2048*128
  if (id < K2_ * HID_) {
    int oc = id & 127, k = id >> 7;
    w2t[id] = (_Float16)w2[oc * K2_ + k];           // w2t[k][oc]
  }
  if (id < HID_ * HID_) {                           // already [k][n] row-major
    wqh[id] = (_Float16)wq[id];
    wkh[id] = (_Float16)wk[id];
    wvh[id] = (_Float16)wv[id];
    woh[id] = (_Float16)wo[id];
  }
}

// ---------------- K1: conv1 direct, ReLU, f16 store ------------------------
__global__ void conv1_kernel(const float* __restrict__ x, const float* __restrict__ w1,
                             const float* __restrict__ b1, _Float16* __restrict__ h1) {
  int idx = blockIdx.x * blockDim.x + threadIdx.x;  // B*128*61*29
  if (idx >= B_ * HID_ * H1Y * H1X) return;
  int xo = idx % H1X; int t = idx / H1X;
  int yo = t % H1Y;  t /= H1Y;
  int oc = t % HID_; int b = t / HID_;
  float acc = b1[oc];
  const float* wp = w1 + oc * NFEAT * 16;
  const float* xp = x + ((size_t)(b * NFEAT) * TLEN + yo) * SCALE_ + xo;
  for (int ic = 0; ic < NFEAT; ++ic)
#pragma unroll
    for (int ky = 0; ky < 4; ++ky)
#pragma unroll
      for (int kx = 0; kx < 4; ++kx)
        acc += xp[ic * TLEN * SCALE_ + ky * SCALE_ + kx] * wp[ic * 16 + ky * 4 + kx];
  h1[idx] = (_Float16)fmaxf(acc, 0.f);
}

// ---------------- K2: conv2 as implicit-im2col WMMA GEMM -------------------
// grid = (24, 1, B), block = 128 (4 waves). Tile: 64 positions x 128 oc.
__global__ __launch_bounds__(128) void conv2_gemm(const _Float16* __restrict__ h1,
                                                  const _Float16* __restrict__ w2t,
                                                  const float* __restrict__ b2,
                                                  _Float16* __restrict__ seq) {
  __shared__ alignas(32) _Float16 sA[64 * 40];     // 64 rows x 32 K (pad 8)
  __shared__ alignas(32) _Float16 sB[32 * 128];    // fragment-order K-chunk
  int b = blockIdx.z;
  int pbase = blockIdx.x * 64;
  int tid = threadIdx.x, lane = tid & 31, wave = tid >> 5;
  v8f acc[8] = {};
  for (int kc = 0; kc < K2_ / 32; ++kc) {
    if (kc + 1 < K2_ / 32)
      __builtin_prefetch(w2t + (size_t)(kc + 1) * 32 * 128 + tid * 8, 0, 1);
    // gather A chunk (im2col on the fly)
#pragma unroll
    for (int i = 0; i < 16; ++i) {
      int e = i * 128 + tid;
      int r = e >> 5, kk = e & 31;
      int p = pbase + r;
      _Float16 v = (_Float16)0.f;
      if (p < L_) {
        int k = kc * 32 + kk;
        int ic = k >> 4, ky = (k >> 2) & 3, kx = k & 3;
        int y = p / H2X + ky, xx = p % H2X + kx;
        v = h1[((b * HID_ + ic) * H1Y + y) * H1X + xx];
      }
      sA[r * 40 + kk] = v;
    }
    // B chunk -> fragment order (8 coalesced u16 loads -> one b128 LDS store)
#pragma unroll
    for (int i = 0; i < 4; ++i) {
      int k0 = i * 8;
      v8h val;
#pragma unroll
      for (int j = 0; j < 8; ++j)
        val[j] = w2t[(size_t)(kc * 32 + k0 + j) * 128 + tid];
      int lanehi = (k0 >> 4) & 1;
      int j0 = k0 & 15;
      int lane_ = (tid & 15) | (lanehi << 4);
      int nb = tid >> 4;
      *(v8h*)(sB + (nb * 32 + lane_) * 16 + j0) = val;
    }
    __syncthreads();
    v16h a = load_a_frag(sA + wave * 16 * 40, 40, lane);
#pragma unroll
    for (int nb = 0; nb < 8; ++nb) {
      v16h bb = load_b_frag(sB, nb, lane);
      acc[nb] = __builtin_amdgcn_wmma_f32_16x16x32_f16(false, a, false, bb,
                                                       (short)0, acc[nb], false, false);
    }
    __syncthreads();
  }
  int n0 = lane & 15;
  int mofs = (lane & 16) ? 8 : 0;
#pragma unroll
  for (int nb = 0; nb < 8; ++nb) {
    int n = nb * 16 + n0;
    float bias = b2[n];
#pragma unroll
    for (int r = 0; r < 8; ++r) {
      int p = pbase + wave * 16 + mofs + r;
      if (p < L_)
        seq[((size_t)b * L_ + p) * HID_ + n] = (_Float16)fmaxf(acc[nb][r] + bias, 0.f);
    }
  }
}

// ---------------- K3: generic (M x 128) x (128 x 128) WMMA GEMM + bias -----
// grid = (M/64), block = 128. f16 A, f16 W (K-major), f32 out.
__global__ __launch_bounds__(128) void gemm128(const _Float16* __restrict__ A,
                                               const _Float16* __restrict__ W,
                                               const float* __restrict__ bias,
                                               float* __restrict__ out, int M) {
  __shared__ alignas(32) _Float16 sA[64 * 136];     // row-major, 16B-aligned rows
  __shared__ alignas(32) _Float16 sW[128 * 128];    // fragment order (kc,nb,lane,j)
  int tid = threadIdx.x, lane = tid & 31, wave = tid >> 5;
  int rbase = blockIdx.x * 64;
  // W -> fragment order
#pragma unroll
  for (int i = 0; i < 16; ++i) {
    int k0 = i * 8;
    v8h val;
#pragma unroll
    for (int j = 0; j < 8; ++j)
      val[j] = W[(size_t)(k0 + j) * 128 + tid];
    int kc = k0 >> 5;
    int lanehi = (k0 >> 4) & 1;
    int j0 = k0 & 15;
    int lane_ = (tid & 15) | (lanehi << 4);
    int nb = tid >> 4;
    *(v8h*)(sW + ((kc * 8 + nb) * 32 + lane_) * 16 + j0) = val;
  }
  // A rows (vectorized 16B loads/stores; 16 threads cover one 256B row)
#pragma unroll
  for (int i = 0; i < 8; ++i) {
    int e = i * 128 + tid;
    int r = e >> 4, cg = e & 15;
    v8h val = {};
    int row = rbase + r;
    if (row < M) val = *(const v8h*)(A + (size_t)row * 128 + cg * 8);
    *(v8h*)(sA + r * 136 + cg * 8) = val;
  }
  __syncthreads();
  v8f acc[8] = {};
#pragma unroll
  for (int kc = 0; kc < 4; ++kc) {
    v16h a = load_a_frag(sA + wave * 16 * 136 + kc * 32, 136, lane);
#pragma unroll
    for (int nb = 0; nb < 8; ++nb) {
      v16h bb = load_b_frag(sW + kc * 8 * 32 * 16, nb, lane);
      acc[nb] = __builtin_amdgcn_wmma_f32_16x16x32_f16(false, a, false, bb,
                                                       (short)0, acc[nb], false, false);
    }
  }
  int n0 = lane & 15;
  int mofs = (lane & 16) ? 8 : 0;
#pragma unroll
  for (int nb = 0; nb < 8; ++nb) {
    int n = nb * 16 + n0;
    float bv = bias[n];
#pragma unroll
    for (int r = 0; r < 8; ++r) {
      int row = rbase + wave * 16 + mofs + r;
      if (row < M) out[(size_t)row * 128 + n] = acc[nb][r] + bv;
    }
  }
}

// ---------------- K4: QK_sample + sparsity metric M ------------------------
// one wave per (b,h,l)
__global__ __launch_bounds__(256) void qk_sample_kernel(const float* __restrict__ Q,
                                                        const float* __restrict__ K,
                                                        const int* __restrict__ idxs,
                                                        float* __restrict__ Mbuf) {
  int gw = (blockIdx.x * blockDim.x + threadIdx.x) >> 5;
  int lane = threadIdx.x & 31;
  if (gw >= B_ * NH_ * L_) return;
  int l = gw % L_; int bh = gw / L_;
  int h = bh % NH_; int b = bh / NH_;
  const float* q = Q + ((size_t)b * L_ + l) * HID_ + h * DH_;
  float mx = -1e30f, sm = 0.f;
  for (int u = lane; u < U_; u += 32) {
    int kidx = idxs[l * U_ + u];
    const float* kp = K + ((size_t)b * L_ + kidx) * HID_ + h * DH_;
    float s = 0.f;
    for (int d = 0; d < DH_; ++d) s += q[d] * kp[d];
    mx = fmaxf(mx, s);
    sm += s;
  }
#pragma unroll
  for (int off = 16; off > 0; off >>= 1) {
    mx = fmaxf(mx, __shfl_xor(mx, off, 32));
    sm += __shfl_xor(sm, off, 32);
  }
  if (lane == 0) Mbuf[gw] = mx - sm / (float)L_;
}

// ---------------- K5: top-U selection per (b,h) ----------------------------
__global__ __launch_bounds__(256) void topk_kernel(const float* __restrict__ Mbuf,
                                                   int* __restrict__ Mtop) {
  __shared__ float vals[L_];
  __shared__ float rv[256];
  __shared__ int   ri[256];
  int bh = blockIdx.x, tid = threadIdx.x;
  for (int i = tid; i < L_; i += 256) vals[i] = Mbuf[bh * L_ + i];
  __syncthreads();
  for (int it = 0; it < U_; ++it) {
    float best = -1e30f; int bi = 0;
    for (int i = tid; i < L_; i += 256)
      if (vals[i] > best) { best = vals[i]; bi = i; }
    rv[tid] = best; ri[tid] = bi;
    __syncthreads();
    for (int s = 128; s > 0; s >>= 1) {
      if (tid < s && rv[tid + s] > rv[tid]) { rv[tid] = rv[tid + s]; ri[tid] = ri[tid + s]; }
      __syncthreads();
    }
    if (tid == 0) { Mtop[bh * U_ + it] = ri[0]; vals[ri[0]] = -1e30f; }
    __syncthreads();
  }
}

// ---------------- K6: dense attention for the U selected queries -----------
// one block per (b,h,u)
__global__ __launch_bounds__(256) void attn_kernel(const float* __restrict__ Q,
                                                   const float* __restrict__ K,
                                                   const float* __restrict__ V,
                                                   const int* __restrict__ Mtop,
                                                   float* __restrict__ upd) {
  __shared__ float q[DH_];
  __shared__ float probs[L_];
  __shared__ float red[256];
  __shared__ float part[4 * DH_];
  int gb = blockIdx.x;
  int u = gb % U_; int bh = gb / U_;
  int h = bh % NH_; int b = bh / NH_;
  int tid = threadIdx.x;
  int l0 = Mtop[bh * U_ + u];
  if (tid < DH_) q[tid] = Q[((size_t)b * L_ + l0) * HID_ + h * DH_ + tid];
  __syncthreads();
  float lmx = -1e30f;
  for (int k = tid; k < L_; k += 256) {
    const float* kp = K + ((size_t)b * L_ + k) * HID_ + h * DH_;
    float s = 0.f;
    for (int d = 0; d < DH_; ++d) s += q[d] * kp[d];
    s *= 0.125f;                      // 1/sqrt(64)
    probs[k] = s;
    lmx = fmaxf(lmx, s);
  }
  red[tid] = lmx; __syncthreads();
  for (int s = 128; s > 0; s >>= 1) {
    if (tid < s) red[tid] = fmaxf(red[tid], red[tid + s]);
    __syncthreads();
  }
  float mx = red[0]; __syncthreads();
  float lsum = 0.f;
  for (int k = tid; k < L_; k += 256) {
    float e = __expf(probs[k] - mx);
    probs[k] = e; lsum += e;
  }
  red[tid] = lsum; __syncthreads();
  for (int s = 128; s > 0; s >>= 1) {
    if (tid < s) red[tid] += red[tid + s];
    __syncthreads();
  }
  float inv = 1.f / red[0];
  int d = tid & 63, pr = tid >> 6;
  float ps = 0.f;
  for (int k = pr; k < L_; k += 4)
    ps += probs[k] * V[((size_t)b * L_ + k) * HID_ + h * DH_ + d];
  part[pr * DH_ + d] = ps;
  __syncthreads();
  if (tid < DH_) {
    float s = part[tid] + part[DH_ + tid] + part[2 * DH_ + tid] + part[3 * DH_ + tid];
    upd[((size_t)bh * U_ + u) * DH_ + tid] = s * inv;
  }
}

// ---------------- K7: V mean, ctx fill, scatter ----------------------------
__global__ void vmean_kernel(const float* __restrict__ V, float* __restrict__ vmean) {
  int idx = blockIdx.x * blockDim.x + threadIdx.x;   // B*NH*DH
  if (idx >= B_ * NH_ * DH_) return;
  int d = idx % DH_; int bh = idx / DH_;
  int h = bh % NH_; int b = bh / NH_;
  float s = 0.f;
  for (int l = 0; l < L_; ++l) s += V[((size_t)b * L_ + l) * HID_ + h * DH_ + d];
  vmean[idx] = s / (float)L_;
}

__global__ void ctx_fill_kernel(const float* __restrict__ vmean, _Float16* __restrict__ ctx) {
  int idx = blockIdx.x * blockDim.x + threadIdx.x;   // B*L*128
  if (idx >= B_ * L_ * HID_) return;
  int c = idx % HID_; int bl = idx / HID_;
  int b = bl / L_;
  int h = c >> 6, d = c & 63;
  ctx[idx] = (_Float16)vmean[(b * NH_ + h) * DH_ + d];
}

__global__ void ctx_scatter_kernel(const float* __restrict__ upd, const int* __restrict__ Mtop,
                                   _Float16* __restrict__ ctx) {
  int idx = blockIdx.x * blockDim.x + threadIdx.x;   // B*NH*U*DH
  if (idx >= B_ * NH_ * U_ * DH_) return;
  int d = idx & 63; int t = idx >> 6;
  int u = t % U_; int bh = t / U_;
  int h = bh % NH_; int b = bh / NH_;
  int l = Mtop[bh * U_ + u];
  ctx[((size_t)b * L_ + l) * HID_ + h * DH_ + d] = (_Float16)upd[idx];
}

// ---------------- launch ---------------------------------------------------
static inline size_t align256(size_t v) { return (v + 255) & ~(size_t)255; }

extern "C" void kernel_launch(void* const* d_in, const int* in_sizes, int n_in,
                              void* d_out, int out_size, void* d_ws, size_t ws_size,
                              hipStream_t stream) {
  const float* x   = (const float*)d_in[0];
  const float* w1  = (const float*)d_in[1];
  const float* b1  = (const float*)d_in[2];
  const float* w2  = (const float*)d_in[3];
  const float* b2  = (const float*)d_in[4];
  const float* wq  = (const float*)d_in[5];
  const float* bq  = (const float*)d_in[6];
  const float* wk  = (const float*)d_in[7];
  const float* bk  = (const float*)d_in[8];
  const float* wv  = (const float*)d_in[9];
  const float* bv  = (const float*)d_in[10];
  const float* wo  = (const float*)d_in[11];
  const float* bo  = (const float*)d_in[12];
  const int*   idx = (const int*)d_in[13];
  float* out = (float*)d_out;

  char* ws = (char*)d_ws;
  size_t off = 0;
  _Float16* w2t = (_Float16*)(ws + off); off = align256(off + (size_t)K2_ * HID_ * 2);
  _Float16* wqh = (_Float16*)(ws + off); off = align256(off + (size_t)HID_ * HID_ * 2);
  _Float16* wkh = (_Float16*)(ws + off); off = align256(off + (size_t)HID_ * HID_ * 2);
  _Float16* wvh = (_Float16*)(ws + off); off = align256(off + (size_t)HID_ * HID_ * 2);
  _Float16* woh = (_Float16*)(ws + off); off = align256(off + (size_t)HID_ * HID_ * 2);
  _Float16* h1  = (_Float16*)(ws + off); off = align256(off + (size_t)B_ * HID_ * H1Y * H1X * 2);
  _Float16* seq = (_Float16*)(ws + off); off = align256(off + (size_t)ML_ * HID_ * 2);
  float* Qb   = (float*)(ws + off); off = align256(off + (size_t)ML_ * HID_ * 4);
  float* Kb   = (float*)(ws + off); off = align256(off + (size_t)ML_ * HID_ * 4);
  float* Vb   = (float*)(ws + off); off = align256(off + (size_t)ML_ * HID_ * 4);
  float* Mbuf = (float*)(ws + off); off = align256(off + (size_t)B_ * NH_ * L_ * 4);
  int*   Mtop = (int*)(ws + off);   off = align256(off + (size_t)B_ * NH_ * U_ * 4);
  float* upd  = (float*)(ws + off); off = align256(off + (size_t)B_ * NH_ * U_ * DH_ * 4);
  float* vmean= (float*)(ws + off); off = align256(off + (size_t)B_ * NH_ * DH_ * 4);
  _Float16* ctx = (_Float16*)(ws + off); off = align256(off + (size_t)ML_ * HID_ * 2);
  (void)ws_size; (void)in_sizes; (void)n_in; (void)out_size;

  prep_weights<<<2048, 128, 0, stream>>>(w2, wq, wk, wv, wo, w2t, wqh, wkh, wvh, woh);

  int n1 = B_ * HID_ * H1Y * H1X;
  conv1_kernel<<<(n1 + 255) / 256, 256, 0, stream>>>(x, w1, b1, h1);

  conv2_gemm<<<dim3(24, 1, B_), 128, 0, stream>>>(h1, w2t, b2, seq);

  gemm128<<<ML_ / 64, 128, 0, stream>>>(seq, wqh, bq, Qb, ML_);
  gemm128<<<ML_ / 64, 128, 0, stream>>>(seq, wkh, bk, Kb, ML_);
  gemm128<<<ML_ / 64, 128, 0, stream>>>(seq, wvh, bv, Vb, ML_);

  qk_sample_kernel<<<(B_ * NH_ * L_) / 8, 256, 0, stream>>>(Qb, Kb, idx, Mbuf);
  topk_kernel<<<B_ * NH_, 256, 0, stream>>>(Mbuf, Mtop);
  attn_kernel<<<B_ * NH_ * U_, 256, 0, stream>>>(Qb, Kb, Vb, Mtop, upd);

  vmean_kernel<<<(B_ * NH_ * DH_ + 255) / 256, 256, 0, stream>>>(Vb, vmean);
  ctx_fill_kernel<<<(B_ * L_ * HID_ + 255) / 256, 256, 0, stream>>>(vmean, ctx);
  ctx_scatter_kernel<<<(B_ * NH_ * U_ * DH_ + 255) / 256, 256, 0, stream>>>(upd, Mtop, ctx);

  gemm128<<<ML_ / 64, 128, 0, stream>>>(ctx, woh, bo, out, ML_);
}
